// Attention_16028817948779
// MI455X (gfx1250) — matrix operand order repaired and checked by reading, not compiled, not run on hardware
//
#include <hip/hip_runtime.h>
#include <hip/hip_bf16.h>
#include <math.h>

typedef __attribute__((ext_vector_type(16))) _Float16 v16h;
typedef __attribute__((ext_vector_type(8)))  float    v8f;

#define BATCH 4
#define CCH   256      // input channels
#define NSP   2304     // 48*48 spatial
#define HEADS 8
#define DHEAD 64
#define HID   512      // HEADS*DHEAD
#define O3    1536     // 3*HID
#define SCALE 10.0f

// padded LDS strides (f16 elements) to spread fragment reads across banks
#define KST 72         // K tile row stride (64 data + 8 pad) -> 144B
#define VST 40         // V tile row stride (32 data + 8 pad) -> 80B

static __device__ __forceinline__ v8f zero8() {
  v8f z;
#pragma unroll
  for (int i = 0; i < 8; i++) z[i] = 0.0f;
  return z;
}

static __device__ __forceinline__ v8f wmma_f16(const v16h& a, const v16h& b, const v8f& c) {
  return __builtin_amdgcn_wmma_f32_16x16x32_f16(false, a, false, b, (short)0, c, false, false);
}

// async DMA: 16 bytes per lane, global -> LDS, tracked by ASYNCcnt
static __device__ __forceinline__ void async_copy_b128(unsigned lds_off, const _Float16* gaddr) {
  asm volatile("global_load_async_to_lds_b128 %0, %1, off"
               :: "v"(lds_off), "v"(gaddr) : "memory");
}
#define WAIT_ASYNC_4() asm volatile("s_wait_asynccnt 0x4" ::: "memory")
#define WAIT_ASYNC_0() asm volatile("s_wait_asynccnt 0x0" ::: "memory")

// ---------------------------------------------------------------------------
// x [b, c, n] f32  ->  xT [b, n, c] f16   (B-fragment-friendly for QKV GEMM)
// ---------------------------------------------------------------------------
__global__ void attn_xpose_kernel(const float* __restrict__ x, _Float16* __restrict__ xt) {
  size_t idx = (size_t)blockIdx.x * 256 + threadIdx.x;   // total BATCH*NSP*CCH
  if (idx >= (size_t)BATCH * NSP * CCH) return;
  int c = (int)(idx & (CCH - 1));
  size_t t = idx >> 8;
  int n = (int)(t % NSP);
  int b = (int)(t / NSP);
  xt[idx] = (_Float16)x[((size_t)b * CCH + c) * NSP + n];
}

// generic f32 -> f16
__global__ void attn_cvt16_kernel(const float* __restrict__ s, _Float16* __restrict__ d, int cnt) {
  int i = blockIdx.x * 256 + threadIdx.x;
  if (i < cnt) d[i] = (_Float16)s[i];
}

// ---------------------------------------------------------------------------
// QKV GEMM:  qkv[b][o][n] = sum_c wq[o][c] * x[b][c][n]
//   one wave computes a 16x64 strip: A-frag amortized over 4 WMMAs per K step
// ---------------------------------------------------------------------------
__global__ __launch_bounds__(128) void attn_qkv_gemm_kernel(
    const _Float16* __restrict__ W,    // [O3][CCH]
    const _Float16* __restrict__ XT,   // [B][NSP][CCH]
    float* __restrict__ QKV)           // [B][O3][NSP]
{
  int lane = threadIdx.x & 31, wave = threadIdx.x >> 5;
  int nbase = blockIdx.x * 64;            // 36 groups of 64 cols
  int otile = blockIdx.y * 4 + wave;      // 0..95
  int b = blockIdx.z;
  int obase = otile * 16;
  int half = lane >> 4, l15 = lane & 15;

  v8f acc[4];
#pragma unroll
  for (int nt = 0; nt < 4; nt++) acc[nt] = zero8();

  for (int kb = 0; kb < CCH; kb += 32) {
    v16h a;
    const _Float16* ar = W + (size_t)(obase + l15) * CCH + kb;
#pragma unroll
    for (int e = 0; e < 8; e++) {
      a[e]     = ar[16 * half + e];
      a[8 + e] = ar[16 * half + 8 + e];
    }
#pragma unroll
    for (int nt = 0; nt < 4; nt++) {
      v16h bf;
      const _Float16* br =
          XT + ((size_t)b * NSP + nbase + nt * 16 + l15) * CCH + kb + 16 * half;
#pragma unroll
      for (int e = 0; e < 16; e++) bf[e] = br[e];
      acc[nt] = wmma_f16(a, bf, acc[nt]);
    }
  }
#pragma unroll
  for (int nt = 0; nt < 4; nt++) {
    float* out = QKV + ((size_t)b * O3 + obase) * NSP + nbase + nt * 16 + l15;
#pragma unroll
    for (int r = 0; r < 8; r++) out[(size_t)(r + 8 * half) * NSP] = acc[nt][r];
  }
}

// ---------------------------------------------------------------------------
// Row L2 norms over n for q (o=0..511) and k (o=512..1023)
// ---------------------------------------------------------------------------
__global__ void attn_rownorm_kernel(const float* __restrict__ QKV, float* __restrict__ norms) {
  int o = blockIdx.x;     // 0..1023
  int b = blockIdx.y;
  const float* row = QKV + ((size_t)b * O3 + o) * NSP;
  float s = 0.0f;
  for (int i = threadIdx.x; i < NSP; i += 256) { float v = row[i]; s += v * v; }
  __shared__ float red[256];
  red[threadIdx.x] = s;
  __syncthreads();
  for (int st = 128; st > 0; st >>= 1) {
    if (threadIdx.x < st) red[threadIdx.x] += red[threadIdx.x + st];
    __syncthreads();
  }
  if (threadIdx.x == 0) norms[b * 1024 + o] = sqrtf(red[0]);
}

// ---------------------------------------------------------------------------
// Normalize + convert:
//   q,k -> [b][H][n][d] f16 (A-frag / K^T B-frag friendly)
//   v   -> [b][H][d][n] f16 (V^T B-frag friendly)
// ---------------------------------------------------------------------------
__global__ void attn_qkn_cvt_kernel(const float* __restrict__ QKV,
                                    const float* __restrict__ norms,
                                    _Float16* __restrict__ QH,
                                    _Float16* __restrict__ KH,
                                    _Float16* __restrict__ VH) {
  size_t idx = (size_t)blockIdx.x * 256 + threadIdx.x;  // total B*O3*NSP
  if (idx >= (size_t)BATCH * O3 * NSP) return;
  int n = (int)(idx % NSP);
  size_t t = idx / NSP;
  int o = (int)(t % O3);
  int b = (int)(t / O3);
  float v = QKV[idx];
  if (o < HID) {
    float nm = fmaxf(norms[b * 1024 + o], 1e-12f);
    int h = o >> 6, d = o & 63;
    QH[(((size_t)b * HEADS + h) * NSP + n) * DHEAD + d] = (_Float16)(v / nm);
  } else if (o < 2 * HID) {
    float nm = fmaxf(norms[b * 1024 + o], 1e-12f);
    int o2 = o - HID;
    int h = o2 >> 6, d = o2 & 63;
    KH[(((size_t)b * HEADS + h) * NSP + n) * DHEAD + d] = (_Float16)(v / nm);
  } else {
    int o3 = o - 2 * HID;
    VH[((size_t)b * HID + o3) * NSP + n] = (_Float16)v;
  }
}

// ---------------------------------------------------------------------------
// Flash attention, async double-buffered K/V staging:
//   - per j-block (32 cols) the whole WG needs K tile 32x64 f16 and V tile
//     64x32 f16; staged ONCE per block via global_load_async_to_lds_b128
//     (4 async instrs/thread), double buffered, s_wait_asynccnt + barrier.
//   - S = Q x K^T (2 WMMAs per 16-col tile), online softmax via half-wave
//     butterflies, P relayout C->A via per-wave LDS tile, out += P x V^T.
// ---------------------------------------------------------------------------
__global__ __launch_bounds__(128) void attn_flash_kernel(
    const _Float16* __restrict__ QH,  // [B][H][N][D]
    const _Float16* __restrict__ KH,  // [B][H][N][D]
    const _Float16* __restrict__ VH,  // [B][H][D][N]
    _Float16* __restrict__ O)         // [B][N][HID]
{
  __shared__ _Float16 kbuf[2][32 * KST];   // 2 * 4608 B
  __shared__ _Float16 vbuf[2][64 * VST];   // 2 * 5120 B
  __shared__ _Float16 stage[4][16][32];    // 4096 B (per-wave P relayout)

  int tid = threadIdx.x;
  int lane = tid & 31, wave = tid >> 5;
  int h = blockIdx.y, b = blockIdx.z;
  int ibase = blockIdx.x * 64 + wave * 16;
  int half = lane >> 4, l15 = lane & 15;

  const _Float16* qbase = QH + (((size_t)b * HEADS + h) * NSP) * DHEAD;
  const _Float16* kbase = KH + (((size_t)b * HEADS + h) * NSP) * DHEAD;
  const _Float16* vbase = VH + (((size_t)b * HEADS + h) * DHEAD) * NSP;

  // Q A-fragments (rows = ibase + l15, K-dim = d, two frags cover d=0..63)
  v16h qf[2];
  {
    const _Float16* qrow = qbase + (size_t)(ibase + l15) * DHEAD;
#pragma unroll
    for (int f = 0; f < 2; f++)
#pragma unroll
      for (int e = 0; e < 8; e++) {
        qf[f][e]     = qrow[f * 32 + 16 * half + e];
        qf[f][8 + e] = qrow[f * 32 + 16 * half + 8 + e];
      }
  }

  v8f ofr[4];
#pragma unroll
  for (int db = 0; db < 4; db++) ofr[db] = zero8();
  float mrow[8], lrow[8];
#pragma unroll
  for (int r = 0; r < 8; r++) { mrow[r] = -1e30f; lrow[r] = 0.0f; }

  // issue async prefetch of a K/V j-block into buffer `buf`
  auto prefetch = [&](int jb, int buf) {
    // K tile: 32 rows x 128B, 8 x 16B chunks per row  -> 2 instrs/thread
#pragma unroll
    for (int t = 0; t < 2; t++) {
      int chunk = tid + t * 128;
      int row = chunk >> 3, c = chunk & 7;
      unsigned lds = (unsigned)(size_t)(&kbuf[buf][row * KST + c * 8]);
      async_copy_b128(lds, kbase + (size_t)(jb + row) * DHEAD + c * 8);
    }
    // V tile: 64 rows x 64B, 4 x 16B chunks per row   -> 2 instrs/thread
#pragma unroll
    for (int t = 0; t < 2; t++) {
      int chunk = tid + t * 128;
      int row = chunk >> 2, c = chunk & 3;
      unsigned lds = (unsigned)(size_t)(&vbuf[buf][row * VST + c * 8]);
      async_copy_b128(lds, vbase + (size_t)row * NSP + jb + c * 8);
    }
  };

  prefetch(0, 0);

  const int NBLK = NSP / 32;  // 72
  for (int it = 0; it < NBLK; ++it) {
    int cur = it & 1;
    if (it + 1 < NBLK) {
      prefetch((it + 1) * 32, cur ^ 1);
      WAIT_ASYNC_4();          // current buffer's 4 async instrs complete
    } else {
      WAIT_ASYNC_0();
    }
    __syncthreads();           // all waves' staging of buffer `cur` visible

    const _Float16* kt = kbuf[cur];
    const _Float16* vt = vbuf[cur];

    v8f s0 = zero8(), s1 = zero8();
#pragma unroll
    for (int f = 0; f < 2; f++) {
      v16h kb0, kb1;
      const _Float16* kr0 = kt + (size_t)l15 * KST + f * 32 + 16 * half;
      const _Float16* kr1 = kt + (size_t)(16 + l15) * KST + f * 32 + 16 * half;
#pragma unroll
      for (int e = 0; e < 16; e++) { kb0[e] = kr0[e]; kb1[e] = kr1[e]; }
      s0 = wmma_f16(qf[f], kb0, s0);
      s1 = wmma_f16(qf[f], kb1, s1);
    }

    // online softmax per row (rows r+8*half live across the 16 lanes of a half)
    float p0[8], p1[8];
#pragma unroll
    for (int r = 0; r < 8; r++) {
      float a0 = s0[r] * SCALE, a1 = s1[r] * SCALE;
      float mx = fmaxf(a0, a1);
#pragma unroll
      for (int m = 1; m < 16; m <<= 1) mx = fmaxf(mx, __shfl_xor(mx, m, 32));
      float mnew = fmaxf(mrow[r], mx);
      float scl  = __expf(mrow[r] - mnew);
      float e0 = __expf(a0 - mnew), e1 = __expf(a1 - mnew);
      float sum = e0 + e1;
#pragma unroll
      for (int m = 1; m < 16; m <<= 1) sum += __shfl_xor(sum, m, 32);
      lrow[r] = lrow[r] * scl + sum;
      mrow[r] = mnew;
#pragma unroll
      for (int db = 0; db < 4; db++) ofr[db][r] *= scl;
      p0[r] = e0; p1[r] = e1;
    }

    // C-layout P -> LDS (per-wave private tile; DS ops in-order per wave)
#pragma unroll
    for (int r = 0; r < 8; r++) {
      stage[wave][r + 8 * half][l15]      = (_Float16)p0[r];
      stage[wave][r + 8 * half][16 + l15] = (_Float16)p1[r];
    }
    // A-layout reload of P (rows = l15, K = j-local 0..31)
    v16h pf;
#pragma unroll
    for (int e = 0; e < 8; e++) {
      pf[e]     = stage[wave][l15][16 * half + e];
      pf[8 + e] = stage[wave][l15][16 * half + 8 + e];
    }

    // out += P(16x32) x V^T(32x16) over 4 d-blocks
#pragma unroll
    for (int db = 0; db < 4; db++) {
      v16h vf;
      const _Float16* vr = vt + (size_t)(db * 16 + l15) * VST + 16 * half;
#pragma unroll
      for (int e = 0; e < 16; e++) vf[e] = vr[e];
      ofr[db] = wmma_f16(pf, vf, ofr[db]);
    }

    __syncthreads();           // all waves done with buffer `cur`
  }

  // epilogue: normalize by l and store f16 into O[b][n][h*64+d]
#pragma unroll
  for (int r = 0; r < 8; r++) {
    float inv = 1.0f / lrow[r];
    int i = ibase + r + 8 * half;
#pragma unroll
    for (int db = 0; db < 4; db++) {
      O[((size_t)b * NSP + i) * HID + h * DHEAD + db * 16 + l15] =
          (_Float16)(ofr[db][r] * inv);
    }
  }
}

// ---------------------------------------------------------------------------
// Output projection: y[b][o][n] = sum_hd w_out[o][hd] * O[b][n][hd] + b_out[o]
//   one wave computes a 16x64 strip (A-frag amortized over 4 WMMAs per K step)
// ---------------------------------------------------------------------------
__global__ __launch_bounds__(128) void attn_out_gemm_kernel(
    const _Float16* __restrict__ WO,   // [CCH][HID]
    const _Float16* __restrict__ Oin,  // [B][NSP][HID]
    const float* __restrict__ bout,    // [CCH]
    float* __restrict__ Y)             // [B][CCH][NSP]
{
  int lane = threadIdx.x & 31, wave = threadIdx.x >> 5;
  int nbase = blockIdx.x * 64;           // 36 groups
  int otile = blockIdx.y * 4 + wave;     // 0..15
  int b = blockIdx.z;
  int obase = otile * 16;
  int half = lane >> 4, l15 = lane & 15;

  v8f acc[4];
#pragma unroll
  for (int nt = 0; nt < 4; nt++) acc[nt] = zero8();

  for (int kb = 0; kb < HID; kb += 32) {
    v16h a;
    const _Float16* ar = WO + (size_t)(obase + l15) * HID + kb;
#pragma unroll
    for (int e = 0; e < 8; e++) {
      a[e]     = ar[16 * half + e];
      a[8 + e] = ar[16 * half + 8 + e];
    }
#pragma unroll
    for (int nt = 0; nt < 4; nt++) {
      v16h bf;
      const _Float16* br =
          Oin + ((size_t)b * NSP + nbase + nt * 16 + l15) * HID + kb + 16 * half;
#pragma unroll
      for (int e = 0; e < 16; e++) bf[e] = br[e];
      acc[nt] = wmma_f16(a, bf, acc[nt]);
    }
  }
#pragma unroll
  for (int nt = 0; nt < 4; nt++) {
#pragma unroll
    for (int r = 0; r < 8; r++) {
      int m = r + 8 * half;
      Y[((size_t)b * CCH + obase + m) * NSP + nbase + nt * 16 + l15] =
          acc[nt][r] + bout[obase + m];
    }
  }
}

// ---------------------------------------------------------------------------
// Workspace layout (bytes, all 256-aligned)
// ---------------------------------------------------------------------------
#define OFF_XT   ((size_t)0)           //  4,718,592  xT f16 [B][N][C]
#define OFF_WQ   ((size_t)4718592)     //    786,432  w_qkv f16
#define OFF_WO   ((size_t)5505024)     //    262,144  w_out f16
#define OFF_QKV  ((size_t)5767168)     // 56,623,104  qkv f32 [B][O3][N]
#define OFF_NRM  ((size_t)62390272)    //     16,384  norms f32 [B][1024]
#define OFF_QH   ((size_t)62406656)    //  9,437,184  q f16 [B][H][N][D]
#define OFF_KH   ((size_t)71843840)    //  9,437,184  k f16 [B][H][N][D]
#define OFF_VH   ((size_t)81281024)    //  9,437,184  v f16 [B][H][D][N]
#define OFF_O16  ((size_t)90718208)    //  9,437,184  O f16 [B][N][HID]

extern "C" void kernel_launch(void* const* d_in, const int* in_sizes, int n_in,
                              void* d_out, int out_size, void* d_ws, size_t ws_size,
                              hipStream_t stream) {
  const float* x     = (const float*)d_in[0];   // [4,256,48,48]
  const float* wqkv  = (const float*)d_in[1];   // [1536,256]
  const float* wout  = (const float*)d_in[2];   // [256,512]
  const float* bout  = (const float*)d_in[3];   // [256]
  float* y = (float*)d_out;                     // [4,256,48,48]

  char* ws = (char*)d_ws;
  _Float16* XT  = (_Float16*)(ws + OFF_XT);
  _Float16* WQ  = (_Float16*)(ws + OFF_WQ);
  _Float16* WO  = (_Float16*)(ws + OFF_WO);
  float*    QKV = (float*)   (ws + OFF_QKV);
  float*    NRM = (float*)   (ws + OFF_NRM);
  _Float16* QH  = (_Float16*)(ws + OFF_QH);
  _Float16* KH  = (_Float16*)(ws + OFF_KH);
  _Float16* VH  = (_Float16*)(ws + OFF_VH);
  _Float16* O16 = (_Float16*)(ws + OFF_O16);

  // 1) layout/precision conversions
  attn_xpose_kernel<<<(BATCH * NSP * CCH) / 256, 256, 0, stream>>>(x, XT);
  attn_cvt16_kernel<<<(O3 * CCH) / 256, 256, 0, stream>>>(wqkv, WQ, O3 * CCH);
  attn_cvt16_kernel<<<(CCH * HID) / 256, 256, 0, stream>>>(wout, WO, CCH * HID);

  // 2) QKV GEMM (WMMA f16 -> f32), 16x64 strip per wave
  attn_qkv_gemm_kernel<<<dim3(NSP / 64, (O3 / 16) / 4, BATCH), 128, 0, stream>>>(WQ, XT, QKV);

  // 3) spatial-axis L2 norms for q,k
  attn_rownorm_kernel<<<dim3(1024, BATCH), 256, 0, stream>>>(QKV, NRM);

  // 4) normalize + pack q,k,v into fragment-friendly f16 layouts
  attn_qkn_cvt_kernel<<<(size_t)(BATCH * O3 * NSP) / 256, 256, 0, stream>>>(QKV, NRM, QH, KH, VH);

  // 5) flash attention (WMMA + async double-buffered K/V staging in LDS)
  attn_flash_kernel<<<dim3(NSP / 64, HEADS, BATCH), 128, 0, stream>>>(QH, KH, VH, O16);

  // 6) output projection + bias, 16x64 strip per wave
  attn_out_gemm_kernel<<<dim3(NSP / 64, (CCH / 16) / 4, BATCH), 128, 0, stream>>>(WO, O16, bout, y);
}
